// POSSMCrossAttention_17635135717771
// MI455X (gfx1250) — compile-verified, matrix operand order checked by reading, not run on hardware
//
#include <hip/hip_runtime.h>
#include <hip/hip_bf16.h>

typedef __bf16 bf16;
typedef __attribute__((ext_vector_type(16))) __bf16 v16b;
typedef __attribute__((ext_vector_type(8)))  __bf16 v8b;
typedef __attribute__((ext_vector_type(8)))  float  v8f;
typedef __attribute__((ext_vector_type(4)))  unsigned int u32x4;
typedef __attribute__((ext_vector_type(8)))  int i32x8;
typedef __attribute__((ext_vector_type(4)))  int i32x4;

// ---------------- constants (match reference) ----------------
constexpr int Bb = 16, Kk = 16, S = 512, E = 512;
constexpr int Ll = 16, H = 8, KVH = 2, Dh = 64, BIN = 50;
constexpr int CH = 32;               // s-chunk rows (TDM tile height)
constexpr int NCH = S / CH;          // 16 chunks
constexpr int MT = CH / 16;          // 2 m-tiles per chunk

// LDS strides (elements). All multiples of 8 so 16-byte vector ops stay aligned.
constexpr int LDSP = E + 8;          // sSpikeB row stride (520)
constexpr int LDK  = 128 + 8;        // sK row stride      (136)
constexpr int LDV  = CH + 8;         // sVT row stride     (40)
constexpr int LDPP = CH + 8;         // sP  row stride     (40)
constexpr int LDO  = 512 + 8;        // sO  row stride     (520)

constexpr size_t SMEM_BYTES =
    (size_t)(2 * CH * E) * 4 +                                       // f32 TDM buffers
    (size_t)(CH * LDSP + CH * LDK + 128 * LDV + H * 16 * LDPP + 16 * LDO) * 2 +
    (size_t)(BIN * Dh * 2 + H * 16) * 4;                             // tables + stats

// ---------------- WMMA helpers ----------------
__device__ __forceinline__ v8f bfwmma(v16b a, v16b b, v8f c) {
  return __builtin_amdgcn_wmma_f32_16x16x32_bf16(false, a, false, b, (short)0, c,
                                                 false, false);
}

// 16x32 bf16 fragment load for A (lane&15 -> m row) or n-major B (lane&15 -> n
// row). Each lane: two contiguous 8-element runs at k-offsets {khalf, khalf+16},
// khalf = (lane>=16)*8 (CDNA5 16-bit operand layout).
__device__ __forceinline__ v16b load_frag(const bf16* base, int stride) {
  const int lane = threadIdx.x & 31;
  const bf16* p = base + (lane & 15) * stride + ((lane >> 4) << 3);
  v8b lo = *(const v8b*)(p);
  v8b hi = *(const v8b*)(p + 16);
  return __builtin_shufflevector(lo, hi, 0, 1, 2, 3, 4, 5, 6, 7, 8, 9, 10, 11,
                                 12, 13, 14, 15);
}

__device__ __forceinline__ void store_c8(bf16* dst, v8f c) {
  v8b o;
#pragma unroll
  for (int i = 0; i < 8; ++i) o[i] = (bf16)c[i];
  *(v8b*)dst = o;
}

// ---------------- Tensor Data Mover: 2D tile (512 x CH f32) -> LDS ----------------
// D# per CDNA5 ISA ch.8: group0 = {count/type/addr}, group1 = {dims/strides}.
// This toolchain (clang-23 / therock) exposes the 6-arg builtin.
__device__ __forceinline__ void tdm_load_chunk(const float* gsrc,
                                               unsigned lds_byte_addr) {
  unsigned long long ga = (unsigned long long)(const void*)gsrc;
  u32x4 g0;
  g0[0] = 1u;                                    // count=1, user descriptor
  g0[1] = lds_byte_addr;                         // LDS destination (bytes)
  g0[2] = (unsigned)(ga & 0xffffffffu);          // global_addr[31:0]
  g0[3] = (unsigned)((ga >> 32) & 0x01ffffffu)   // global_addr[56:32]
          | 0x80000000u;                         // type = 2 ("image")
  i32x8 g1;
  g1[0] = 0x20000;                               // data_size = 2 (4-byte elems)
  g1[1] = (int)(512u << 16);                     // tensor_dim0 = 512 (bits 79:48)
  g1[2] = (int)((unsigned)CH << 16);             // tensor_dim1 = CH (bits 111:80)
  g1[3] = (int)(512u << 16);                     // tile_dim0 = 512 (bits 127:112)
  g1[4] = CH;                                    // tile_dim1 = CH, tile_dim2 = 0
  g1[5] = 512;                                   // tensor_dim0_stride = 512
  g1[6] = 0;
  g1[7] = 0;
  i32x4 gz4 = {0, 0, 0, 0};                      // groups 2/3 unused (2D tile)
  i32x8 gz8 = {0, 0, 0, 0, 0, 0, 0, 0};
  __builtin_amdgcn_tensor_load_to_lds(g0, g1, gz4, gz4, gz8, 0);
}

// ---------------- prep kernels ----------------
__global__ void possm_prep_q(const float* __restrict__ lq,
                             const float* __restrict__ Wq,
                             bf16* __restrict__ qb) {
  int idx = blockIdx.x * blockDim.x + threadIdx.x;
  if (idx >= Ll * H * Dh) return;
  int l = idx / (H * Dh);
  int hd = idx - l * (H * Dh);
  float acc = 0.f;
  for (int e = 0; e < E; ++e) acc += lq[l * E + e] * Wq[hd * E + e];
  int h = hd / Dh, d = hd - h * Dh;
  qb[h * (Ll * Dh) + l * Dh + d] = (bf16)(acc * 0.125f);  // fold 1/sqrt(D)
}

__global__ void possm_cast(const float* __restrict__ src, bf16* __restrict__ dst,
                           int n) {
  for (int i = blockIdx.x * blockDim.x + threadIdx.x; i < n;
       i += gridDim.x * blockDim.x)
    dst[i] = (bf16)src[i];
}

// ---------------- fused main kernel: one workgroup per (b,k) ----------------
__global__ __launch_bounds__(256, 1) void possm_main(
    const float* __restrict__ spike, const int* __restrict__ offsets,
    const int* __restrict__ lengths, const float* __restrict__ cosT,
    const float* __restrict__ sinT, const bf16* __restrict__ qb,
    const bf16* __restrict__ WkB, const bf16* __restrict__ WvB,
    const bf16* __restrict__ WoB, float* __restrict__ out) {
  extern __shared__ __align__(16) char smem[];
  float* sF32   = (float*)smem;                   // [2][CH*E] TDM staging (f32)
  bf16* sSpikeB = (bf16*)(sF32 + 2 * CH * E);     // [CH][LDSP] spike chunk (bf16)
  bf16* sK      = sSpikeB + CH * LDSP;            // [CH][LDK]  rope'd keys  [s][dk]
  bf16* sVT     = sK + CH * LDK;                  // [128][LDV] values       [dk][s]
  bf16* sP      = sVT + 128 * LDV;                // [H*16][LDPP] probs      [h][l][s]
  bf16* sO      = sP + H * 16 * LDPP;             // [16][LDO]  attn out     [l][hd]
  float* sCos   = (float*)(sO + 16 * LDO);        // [BIN*Dh]
  float* sSin   = sCos + BIN * Dh;
  float* sStat  = sSin + BIN * Dh;                // [H][16] alpha / inv-sum

  const int bk   = blockIdx.x;
  const int tid  = threadIdx.x;
  const int lane = tid & 31;
  const int w    = tid >> 5;        // wave id == head id
  const int hi   = lane >> 4;
  const int ln   = lane & 15;
  const int kvh  = w >> 2;          // interleaved repeat_kv
  const int len  = lengths[bk];

  for (int i = tid; i < BIN * Dh; i += 256) {
    sCos[i] = cosT[i];
    sSin[i] = sinT[i];
  }

  const long sbase = (long)bk * S * E;
  const unsigned ldsF32 = (unsigned)(unsigned long long)(uintptr_t)(void*)sF32;

  // prime the double buffer: chunks 0 and 1 in flight via TDM (wave 0 only)
  if (w == 0) {
    tdm_load_chunk(spike + sbase, ldsF32);
    tdm_load_chunk(spike + sbase + CH * E, ldsF32 + CH * E * 4);
    __builtin_amdgcn_s_wait_tensorcnt(1);   // chunk 0 landed (in-order)
  }
  __syncthreads();

  v16b qf0 = load_frag(qb + w * (Ll * Dh) + 0, Dh);
  v16b qf1 = load_frag(qb + w * (Ll * Dh) + 32, Dh);

  const v8f vzero = {0.f, 0.f, 0.f, 0.f, 0.f, 0.f, 0.f, 0.f};
  float m_run = -1e30f, s_run = 0.f;
  v8f od[4];
#pragma unroll
  for (int t = 0; t < 4; ++t) od[t] = vzero;

  for (int c = 0; c < NCH; ++c) {
    // ---- LDS f32 -> LDS bf16 convert (flat float8 groups: conflict-free) ----
    {
      const float* src = sF32 + (c & 1) * (CH * E);
#pragma unroll
      for (int j = 0; j < 8; ++j) {
        int flat = (tid + j * 256) * 8;           // 0 .. CH*E-8
        int row = flat >> 9;
        int col = flat & 511;
        float4 f0 = *(const float4*)(src + flat);
        float4 f1 = *(const float4*)(src + flat + 4);
        v8b o;
        o[0] = (bf16)f0.x; o[1] = (bf16)f0.y; o[2] = (bf16)f0.z; o[3] = (bf16)f0.w;
        o[4] = (bf16)f1.x; o[5] = (bf16)f1.y; o[6] = (bf16)f1.z; o[7] = (bf16)f1.w;
        *(v8b*)(&sSpikeB[row * LDSP + col]) = o;
      }
    }
    __syncthreads();

    // buffer (c&1) is free again: DMA chunk c+2 into it while we compute
    if (w == 0 && c + 2 < NCH)
      tdm_load_chunk(spike + sbase + (long)(c + 2) * CH * E,
                     ldsF32 + (unsigned)(c & 1) * CH * E * 4);

    // ---- K and V projections: wave w owns output dims [w*16, w*16+16) ----
    {
      const bf16* wkb = WkB + (w * 16) * E;  // natural [n][e] == B-frag layout
      const bf16* wvb = WvB + (w * 16) * E;
#pragma unroll
      for (int mt = 0; mt < MT; ++mt) {
        v8f ak = vzero, av = vzero;
#pragma unroll
        for (int ks = 0; ks < 16; ++ks) {
          v16b a   = load_frag(&sSpikeB[(mt * 16) * LDSP + ks * 32], LDSP);
          v16b bkf = load_frag(wkb + ks * 32, E);
          v16b bvf = load_frag(wvb + ks * 32, E);
          ak = bfwmma(a, bkf, ak);
          av = bfwmma(a, bvf, av);
        }
#pragma unroll
        for (int i = 0; i < 8; ++i)
          sK[(mt * 16 + i + 8 * hi) * LDK + w * 16 + ln] = (bf16)ak[i];
        store_c8(&sVT[(w * 16 + ln) * LDV + mt * 16 + 8 * hi], av);
      }
    }
    __syncthreads();

    // ---- in-place RoPE on sK ----
    {
      const int* offp = offsets + bk * S + c * CH;
#pragma unroll
      for (int jj = 0; jj < (CH * 64) / 256; ++jj) {
        int pid = tid + jj * 256;
        int s  = pid >> 6;
        int q6 = pid & 63;
        int kv = q6 >> 5;
        int d  = q6 & 31;
        int off = offp[s];
        float c0 = sCos[off * Dh + d],      s0 = sSin[off * Dh + d];
        float c1 = sCos[off * Dh + d + 32], s1 = sSin[off * Dh + d + 32];
        bf16* p0 = &sK[s * LDK + kv * 64 + d];
        bf16* p1 = p0 + 32;
        float x0 = (float)*p0, x1 = (float)*p1;
        *p0 = (bf16)(x0 * c0 - x1 * s0);
        *p1 = (bf16)(x1 * c1 + x0 * s1);
      }
    }
    __syncthreads();

    // ---- scores^T, online softmax, P @ V (wave == head) ----
    {
      v8f st[MT];
#pragma unroll
      for (int t = 0; t < MT; ++t) {
        st[t] = vzero;
        v16b a0 = load_frag(&sK[(t * 16) * LDK + kvh * 64 + 0], LDK);
        v16b a1 = load_frag(&sK[(t * 16) * LDK + kvh * 64 + 32], LDK);
        st[t] = bfwmma(a0, qf0, st[t]);   // C[m=s][n=l] += K[s,d] * qT[d,l]
        st[t] = bfwmma(a1, qf1, st[t]);
      }
      float cm = -1e30f;
#pragma unroll
      for (int t = 0; t < MT; ++t)
#pragma unroll
        for (int i = 0; i < 8; ++i) {
          int sg = c * CH + t * 16 + i + 8 * hi;
          if (sg >= len) st[t][i] = -1e30f;
          cm = fmaxf(cm, st[t][i]);
        }
      cm = fmaxf(cm, __shfl_xor(cm, 16));
      float m_new = fmaxf(m_run, cm);
      float alpha = exp2f((m_run - m_new) * 1.44269504f);
      float ps = 0.f;
#pragma unroll
      for (int t = 0; t < MT; ++t)
#pragma unroll
        for (int i = 0; i < 8; ++i) {
          float p = exp2f((st[t][i] - m_new) * 1.44269504f);
          st[t][i] = p;
          ps += p;
        }
      ps += __shfl_xor(ps, 16);
      s_run = s_run * alpha + ps;
      m_run = m_new;
      if (lane < 16) sStat[w * 16 + lane] = alpha;
#pragma unroll
      for (int t = 0; t < MT; ++t)
        store_c8(&sP[(w * 16 + ln) * LDPP + t * 16 + 8 * hi], st[t]);
      float av8[8];
#pragma unroll
      for (int i = 0; i < 8; ++i) av8[i] = sStat[w * 16 + i + 8 * hi];
#pragma unroll
      for (int t = 0; t < 4; ++t)
#pragma unroll
        for (int i = 0; i < 8; ++i) od[t][i] *= av8[i];
      v16b pa = load_frag(&sP[(w * 16) * LDPP], LDPP);   // K = CH = 32
#pragma unroll
      for (int dt = 0; dt < 4; ++dt) {
        v16b vb = load_frag(&sVT[(kvh * 64 + dt * 16) * LDV], LDV);
        od[dt] = bfwmma(pa, vb, od[dt]);
      }
    }
    __syncthreads();

    // flip: chunk c+1 must be resident before next convert pass
    if (w == 0 && c + 1 < NCH) {
      if (c + 2 < NCH) __builtin_amdgcn_s_wait_tensorcnt(1);
      else             __builtin_amdgcn_s_wait_tensorcnt(0);
    }
    __syncthreads();
  }

  // ---- normalize and gather heads into sO[l][h*64+d] ----
  if (lane < 16) sStat[w * 16 + lane] = 1.0f / s_run;
  float rs[8];
#pragma unroll
  for (int i = 0; i < 8; ++i) rs[i] = sStat[w * 16 + i + 8 * hi];
#pragma unroll
  for (int dt = 0; dt < 4; ++dt)
#pragma unroll
    for (int i = 0; i < 8; ++i) {
      float v = od[dt][i] * rs[i];
      sO[(i + 8 * hi) * LDO + w * 64 + dt * 16 + ln] = (bf16)v;
    }
  __syncthreads();

  // ---- final projection: out[l][e] = sO[l][hd] @ Wo[e][hd] ----
  const long obase = (long)bk * Ll * E;
#pragma unroll
  for (int jt = 0; jt < 4; ++jt) {
    int nt = w + jt * 8;
    v8f co = vzero;
#pragma unroll
    for (int ks = 0; ks < 16; ++ks) {
      v16b a = load_frag(&sO[ks * 32], LDO);
      v16b b = load_frag(WoB + (nt * 16) * 512 + ks * 32, 512);
      co = bfwmma(a, b, co);
    }
#pragma unroll
    for (int i = 0; i < 8; ++i)
      out[obase + (i + 8 * hi) * E + nt * 16 + ln] = co[i];
  }
}

// ---------------- launcher ----------------
extern "C" void kernel_launch(void* const* d_in, const int* in_sizes, int n_in,
                              void* d_out, int out_size, void* d_ws,
                              size_t ws_size, hipStream_t stream) {
  const float* spike   = (const float*)d_in[0];
  const int*   offsets = (const int*)d_in[1];
  const int*   lengths = (const int*)d_in[2];
  const float* cosT    = (const float*)d_in[3];
  const float* sinT    = (const float*)d_in[4];
  const float* lq      = (const float*)d_in[5];
  const float* Wq      = (const float*)d_in[6];
  const float* Wk      = (const float*)d_in[7];
  const float* Wv      = (const float*)d_in[8];
  const float* Wo      = (const float*)d_in[9];
  float* out = (float*)d_out;

  bf16* qb  = (bf16*)d_ws;                 // 16 KB
  bf16* WkB = qb + Ll * H * Dh;            // 128 KB
  bf16* WvB = WkB + (KVH * Dh) * E;        // 128 KB
  bf16* WoB = WvB + (KVH * Dh) * E;        // 512 KB

  possm_prep_q<<<(Ll * H * Dh + 255) / 256, 256, 0, stream>>>(lq, Wq, qb);
  possm_cast<<<256, 256, 0, stream>>>(Wk, WkB, (KVH * Dh) * E);
  possm_cast<<<256, 256, 0, stream>>>(Wv, WvB, (KVH * Dh) * E);
  possm_cast<<<1024, 256, 0, stream>>>(Wo, WoB, E * (H * Dh));

  (void)hipFuncSetAttribute((const void*)possm_main,
                            hipFuncAttributeMaxDynamicSharedMemorySize,
                            (int)SMEM_BYTES);
  possm_main<<<Bb * Kk, 256, SMEM_BYTES, stream>>>(
      spike, offsets, lengths, cosT, sinT, qb, WkB, WvB, WoB, out);
}
